// GPT_40243843563694
// MI455X (gfx1250) — matrix operand order repaired and checked by reading, not compiled, not run on hardware
//
#include <hip/hip_runtime.h>
#include <hip/hip_bf16.h>
#include <math.h>

// ---------------------------------------------------------------------------
// GPT block forward for MI455X (gfx1250): wave32, WMMA f32_16x16x32_f16,
// async global->LDS DMA (ASYNCcnt) double-buffered GEMM tiles.
// L=2, H=16, B=2, T=2048, C=1024, hd=64.
// ---------------------------------------------------------------------------

#define LNUM 2
#define HNUM 16
#define BNUM 2
#define TNUM 2048
#define CNUM 1024
#define HD   64
#define BT   (BNUM * TNUM)

typedef __attribute__((ext_vector_type(16))) _Float16 v16h;
typedef __attribute__((ext_vector_type(8)))  _Float16 v8h;
typedef __attribute__((ext_vector_type(8)))  float    v8f;

// low 32 bits of a flat shared-memory pointer == LDS byte address (aperture rule)
__device__ __forceinline__ unsigned lds_addr32(const void* p) {
    return (unsigned)(unsigned long long)p;
}

// GLOBAL_LOAD_ASYNC_TO_LDS_B128, GVS mode: mem = SGPR_base + VGPR_I32 offset.
// Per-lane 16B transfer into LDS at the lane's VDST address. Tracked by ASYNCcnt.
__device__ __forceinline__ void async_ld_b128(unsigned lds_byte, unsigned byte_off,
                                              unsigned long long base) {
    asm volatile("global_load_async_to_lds_b128 %0, %1, %2"
                 :: "v"(lds_byte), "v"(byte_off), "s"(base)
                 : "memory");
}

__device__ __forceinline__ void wait_async0() {
    asm volatile("s_wait_asynccnt 0" ::: "memory");
}

// -------------------------------------------------------------------- convert
__global__ __launch_bounds__(256) void k_cvt_f32_f16(const float* __restrict__ s,
                                                     _Float16* __restrict__ d, int n) {
    for (int i = blockIdx.x * blockDim.x + threadIdx.x; i < n; i += gridDim.x * blockDim.x)
        d[i] = (_Float16)s[i];
}

// ------------------------------------------------------------------ layernorm
// one block per row, C = 1024, 256 threads -> 4 elems/thread
__global__ __launch_bounds__(256) void k_layernorm(const float* __restrict__ x,
                                                   const float* __restrict__ g,
                                                   const float* __restrict__ b,
                                                   _Float16* __restrict__ out16,
                                                   float* __restrict__ out32) {
    __shared__ float red[256];
    const int row = blockIdx.x;
    const int tid = threadIdx.x;
    const float* xr = x + (size_t)row * CNUM;

    float v0 = xr[tid], v1 = xr[tid + 256], v2 = xr[tid + 512], v3 = xr[tid + 768];

    float s = v0 + v1 + v2 + v3;
    red[tid] = s; __syncthreads();
    for (int o = 128; o > 0; o >>= 1) { if (tid < o) red[tid] += red[tid + o]; __syncthreads(); }
    const float mu = red[0] * (1.0f / CNUM);
    __syncthreads();

    float d0 = v0 - mu, d1 = v1 - mu, d2 = v2 - mu, d3 = v3 - mu;
    red[tid] = d0 * d0 + d1 * d1 + d2 * d2 + d3 * d3; __syncthreads();
    for (int o = 128; o > 0; o >>= 1) { if (tid < o) red[tid] += red[tid + o]; __syncthreads(); }
    const float rstd = rsqrtf(red[0] * (1.0f / CNUM) + 1e-5f);

    #pragma unroll
    for (int j = 0; j < 4; ++j) {
        int c = tid + j * 256;
        float val = (xr[c] - mu) * rstd * g[c] + b[c];
        if (out16) out16[(size_t)row * CNUM + c] = (_Float16)val;
        else       out32[(size_t)row * CNUM + c] = val;
    }
}

// ----------------------------------------------------------------------- gemm
// Y[M,N] = epilogue(A[M,K] @ W[K,N] + bias).  BM=128 BN=128 BK=32.
// 8 waves as 4(M) x 2(N); wave computes 32x64 via 2x4 wmma f32_16x16x32_f16.
// Double-buffered LDS tiles filled by async global->LDS DMA.
#define GEMM_MODE_F16    0   // out16 = acc + bias
#define GEMM_MODE_GELU   1   // out16 = gelu(acc + bias)
#define GEMM_MODE_RESID  2   // out32 = resid + acc + bias

#define LDA_S 40    // As row stride in halves (80B, 16B aligned)
#define LDB_S 144   // Bs row stride in halves (288B, 16B aligned)

__global__ __launch_bounds__(256) void k_gemm(const _Float16* __restrict__ A,
                                              const _Float16* __restrict__ W,
                                              const float* __restrict__ bias,
                                              _Float16* __restrict__ out16,
                                              float* __restrict__ out32,
                                              const float* __restrict__ resid,
                                              int M, int N, int K, int mode) {
    __shared__ _Float16 As[2][128 * LDA_S];
    __shared__ _Float16 Bs[2][32 * LDB_S];

    const int tid  = threadIdx.x;
    const int lane = tid & 31;
    const int wid  = tid >> 5;          // 0..7
    const int waveM = wid & 3;          // 4 waves along M
    const int waveN = wid >> 2;         // 2 waves along N
    const int hi16  = lane >> 4;        // half-wave id
    const int l16   = lane & 15;

    const int m0 = blockIdx.y * 128;
    const int n0 = blockIdx.x * 128;

    const v8f zero8 = {0.f,0.f,0.f,0.f,0.f,0.f,0.f,0.f};
    v8f acc[2][4];
    #pragma unroll
    for (int mi = 0; mi < 2; ++mi)
        #pragma unroll
        for (int ni = 0; ni < 4; ++ni) acc[mi][ni] = zero8;

    // cooperative DMA mappings (each thread owns 32B of each tile)
    const int arow = tid >> 1, acol = (tid & 1) * 16;   // A: 128 x 32
    const int brow = tid >> 3, bcol = (tid & 7) * 16;   // B:  32 x 128

    const unsigned long long baseA = (unsigned long long)A;
    const unsigned long long baseW = (unsigned long long)W;

    auto issue_tile = [&](int k0, int buf) {
        const unsigned la = lds_addr32(&As[buf][arow * LDA_S + acol]);
        const unsigned oa = (unsigned)(((size_t)(m0 + arow) * K + k0 + acol) * 2);
        async_ld_b128(la,      oa,      baseA);
        async_ld_b128(la + 16, oa + 16, baseA);
        const unsigned lb = lds_addr32(&Bs[buf][brow * LDB_S + bcol]);
        const unsigned ob = (unsigned)(((size_t)(k0 + brow) * N + n0 + bcol) * 2);
        async_ld_b128(lb,      ob,      baseW);
        async_ld_b128(lb + 16, ob + 16, baseW);
    };

    issue_tile(0, 0);
    wait_async0();
    __syncthreads();

    int buf = 0;
    for (int k0 = 0; k0 < K; k0 += 32) {
        // DMA next tile while this one feeds the matrix pipes
        if (k0 + 32 < K) issue_tile(k0 + 32, buf ^ 1);

        v16h af[2];
        #pragma unroll
        for (int mi = 0; mi < 2; ++mi) {
            const int r = waveM * 32 + mi * 16 + l16;   // A-matrix row = lane 0..15
            ((v8h*)&af[mi])[0] = *(const v8h*)&As[buf][r * LDA_S + hi16 * 8];
            ((v8h*)&af[mi])[1] = *(const v8h*)&As[buf][r * LDA_S + hi16 * 8 + 16];
        }
        v16h bf[4];
        #pragma unroll
        for (int ni = 0; ni < 4; ++ni) {
            const int noff = waveN * 64 + ni * 16;      // B: n = l16, 16 k in table order
            #pragma unroll
            for (int e = 0; e < 16; ++e)
                bf[ni][e] = Bs[buf][((e < 8) ? hi16 * 8 + e : 16 + hi16 * 8 + e - 8) * LDB_S + noff + l16];
        }
        #pragma unroll
        for (int mi = 0; mi < 2; ++mi)
            #pragma unroll
            for (int ni = 0; ni < 4; ++ni)
                acc[mi][ni] = __builtin_amdgcn_wmma_f32_16x16x32_f16(
                    false, af[mi], false, bf[ni], (short)0, acc[mi][ni], false, false);

        wait_async0();      // next tile resident
        __syncthreads();
        buf ^= 1;
    }

    // epilogue: D row m = v + 8*hi16, col n = l16
    #pragma unroll
    for (int mi = 0; mi < 2; ++mi) {
        #pragma unroll
        for (int ni = 0; ni < 4; ++ni) {
            #pragma unroll
            for (int v = 0; v < 8; ++v) {
                const int gm = m0 + waveM * 32 + mi * 16 + v + 8 * hi16;
                const int gn = n0 + waveN * 64 + ni * 16 + l16;
                float val = acc[mi][ni][v] + bias[gn];
                const size_t idx = (size_t)gm * N + gn;
                if (mode == GEMM_MODE_F16) {
                    out16[idx] = (_Float16)val;
                } else if (mode == GEMM_MODE_GELU) {
                    val = 0.5f * val * (1.0f + erff(val * 0.70710678118654752f));
                    out16[idx] = (_Float16)val;
                } else {
                    out32[idx] = resid[idx] + val;
                }
            }
        }
    }
}

// ------------------------------------------------------------------ attention
// Flash-style. Block = 128 q rows of one (b,h); 8 waves x 16 q rows.
// Per 32-key chunk: Kt (transposed) + V (async DMA) staged in LDS, scores via
// WMMA, online softmax with shfl row reductions, P restaged via LDS, P@V WMMA.
__global__ __launch_bounds__(256) void k_attention(const _Float16* __restrict__ Q,
                                                   const _Float16* __restrict__ Kp,
                                                   const _Float16* __restrict__ Vp,
                                                   _Float16* __restrict__ Y) {
    __shared__ _Float16 Kt[64 * 32];        // [d][j]   4 KB
    __shared__ _Float16 Vs[32 * 64];        // [j][d]   4 KB
    __shared__ _Float16 Ps[8][16 * 32];     // per-wave P scratch, 8 KB

    const int nqb = TNUM / 128;
    const int qb = blockIdx.x % nqb;
    const int h  = (blockIdx.x / nqb) % HNUM;
    const int bb = blockIdx.x / (nqb * HNUM);

    const int tid  = threadIdx.x;
    const int lane = tid & 31;
    const int wid  = tid >> 5;
    const int hi16 = lane >> 4;
    const int l16  = lane & 15;

    const int q0 = qb * 128 + wid * 16;     // this wave's first q row
    const float scale = 0.125f;             // 1/sqrt(64)

    // Q A-fragments (held in registers for whole kernel): rows q0..q0+15, d 0..63
    v16h qf[2];
    {
        const size_t qbase = ((size_t)bb * TNUM + (q0 + l16)) * CNUM + h * HD;
        #pragma unroll
        for (int s = 0; s < 2; ++s) {
            ((v8h*)&qf[s])[0] = *(const v8h*)(Q + qbase + s * 32 + hi16 * 8);
            ((v8h*)&qf[s])[1] = *(const v8h*)(Q + qbase + s * 32 + hi16 * 8 + 16);
        }
    }

    const v8f zero8 = {0.f,0.f,0.f,0.f,0.f,0.f,0.f,0.f};
    v8f oacc[4];
    #pragma unroll
    for (int t = 0; t < 4; ++t) oacc[t] = zero8;
    float m_s[8], l_s[8];
    #pragma unroll
    for (int v = 0; v < 8; ++v) { m_s[v] = -1e30f; l_s[v] = 0.0f; }

    // K/V cooperative-load mappings
    const int kkr   = tid & 31;             // key row within chunk
    const int dbase = (tid >> 5) * 8;       // 8 d per thread
    const int vrow  = tid >> 3;             // 0..31
    const int vcb   = (tid & 7) * 8;

    const unsigned long long baseV = (unsigned long long)Vp;

    for (int k0 = 0; k0 < TNUM; k0 += 32) {
        __syncthreads();
        {   // V tile via async global->LDS DMA (16B per lane)
            const unsigned lv = lds_addr32(&Vs[vrow * 64 + vcb]);
            const unsigned ov =
                (unsigned)((((size_t)bb * TNUM + k0 + vrow) * CNUM + h * HD + vcb) * 2);
            async_ld_b128(lv, ov, baseV);
            // K tile, transposed into Kt[d][j] (element scatter -> VGPR path)
            v8h kv = *(const v8h*)(Kp + ((size_t)bb * TNUM + k0 + kkr) * CNUM + h * HD + dbase);
            #pragma unroll
            for (int j = 0; j < 8; ++j) Kt[(dbase + j) * 32 + kkr] = kv[j];
            if (k0 + 32 < TNUM)  // hint next K chunk toward the caches
                __builtin_prefetch(Kp + ((size_t)bb * TNUM + k0 + 32 + kkr) * CNUM + h * HD + dbase, 0, 3);
        }
        wait_async0();
        __syncthreads();

        // scores: S[16q x 32k] as two 16x16 wmma accumulators
        v8f sa0 = zero8, sa1 = zero8;
        #pragma unroll
        for (int s = 0; s < 2; ++s) {       // d-slices 0..31 / 32..63
            v16h kf0, kf1;
            #pragma unroll
            for (int e = 0; e < 16; ++e) {
                const int kd = s * 32 + ((e < 8) ? hi16 * 8 + e : 16 + hi16 * 8 + e - 8);
                kf0[e] = Kt[kd * 32 + l16];
                kf1[e] = Kt[kd * 32 + 16 + l16];
            }
            sa0 = __builtin_amdgcn_wmma_f32_16x16x32_f16(false, qf[s], false, kf0,
                                                         (short)0, sa0, false, false);
            sa1 = __builtin_amdgcn_wmma_f32_16x16x32_f16(false, qf[s], false, kf1,
                                                         (short)0, sa1, false, false);
        }

        // online softmax per score row (row = v + 8*hi16; 16 cols live in 16 lanes)
        _Float16* Pw = &Ps[wid][0];
        #pragma unroll
        for (int v = 0; v < 8; ++v) {
            const int qg = q0 + v + 8 * hi16;
            const int kg0 = k0 + l16, kg1 = k0 + 16 + l16;
            float s0 = sa0[v] * scale, s1 = sa1[v] * scale;
            const bool causal = (qg % 3) == 0;
            if (causal && kg0 > qg) s0 = -1e30f;
            if (causal && kg1 > qg) s1 = -1e30f;
            float rm = fmaxf(s0, s1);
            rm = fmaxf(rm, __shfl_xor(rm, 1, 32));
            rm = fmaxf(rm, __shfl_xor(rm, 2, 32));
            rm = fmaxf(rm, __shfl_xor(rm, 4, 32));
            rm = fmaxf(rm, __shfl_xor(rm, 8, 32));
            const float mn = fmaxf(m_s[v], rm);
            const float alpha = __expf(m_s[v] - mn);
            const float p0 = __expf(s0 - mn), p1 = __expf(s1 - mn);
            float rs = p0 + p1;
            rs += __shfl_xor(rs, 1, 32);
            rs += __shfl_xor(rs, 2, 32);
            rs += __shfl_xor(rs, 4, 32);
            rs += __shfl_xor(rs, 8, 32);
            m_s[v] = mn;
            l_s[v] = l_s[v] * alpha + rs;
            #pragma unroll
            for (int t = 0; t < 4; ++t) oacc[t][v] *= alpha;
            const int r = v + 8 * hi16;
            Pw[r * 32 + l16]      = (_Float16)p0;
            Pw[r * 32 + 16 + l16] = (_Float16)p1;
        }
        asm volatile("" ::: "memory");      // same-wave LDS store->load ordering

        // P fragment (A layout) from LDS, then P @ V
        v16h pf;
        ((v8h*)&pf)[0] = *(const v8h*)&Pw[l16 * 32 + hi16 * 8];
        ((v8h*)&pf)[1] = *(const v8h*)&Pw[l16 * 32 + hi16 * 8 + 16];
        #pragma unroll
        for (int t = 0; t < 4; ++t) {
            v16h vf;
            #pragma unroll
            for (int e = 0; e < 16; ++e)
                vf[e] = Vs[((e < 8) ? hi16 * 8 + e : 16 + hi16 * 8 + e - 8) * 64 + t * 16 + l16];
            oacc[t] = __builtin_amdgcn_wmma_f32_16x16x32_f16(false, pf, false, vf,
                                                             (short)0, oacc[t], false, false);
        }
    }

    // normalize + write y (f16, [B,T,C] head-concat layout)
    #pragma unroll
    for (int v = 0; v < 8; ++v) {
        const float rinv = 1.0f / l_s[v];
        const int qg = q0 + v + 8 * hi16;
        const size_t base = ((size_t)bb * TNUM + qg) * CNUM + h * HD;
        #pragma unroll
        for (int t = 0; t < 4; ++t)
            Y[base + t * 16 + l16] = (_Float16)(oacc[t][v] * rinv);
    }
}

// ---------------------------------------------------------------------- host
extern "C" void kernel_launch(void* const* d_in, const int* in_sizes, int n_in,
                              void* d_out, int out_size, void* d_ws, size_t ws_size,
                              hipStream_t stream) {
    (void)in_sizes; (void)n_in; (void)out_size; (void)ws_size;

    const float* x     = (const float*)d_in[0];
    const float* ln1_g = (const float*)d_in[1];
    const float* ln1_b = (const float*)d_in[2];
    const float* wq    = (const float*)d_in[3];
    const float* bq    = (const float*)d_in[4];
    const float* wk    = (const float*)d_in[5];
    const float* bk    = (const float*)d_in[6];
    const float* wv    = (const float*)d_in[7];
    const float* bv    = (const float*)d_in[8];
    const float* wo    = (const float*)d_in[9];
    const float* bo    = (const float*)d_in[10];
    const float* ln2_g = (const float*)d_in[11];
    const float* ln2_b = (const float*)d_in[12];
    const float* w1    = (const float*)d_in[13];
    const float* b1    = (const float*)d_in[14];
    const float* w2    = (const float*)d_in[15];
    const float* b2    = (const float*)d_in[16];
    const float* lnf_g = (const float*)d_in[17];
    const float* lnf_b = (const float*)d_in[18];

    // workspace carve-up (256B aligned)
    char* ws = (char*)d_ws;
    size_t off = 0;
    auto take = [&](size_t bytes) -> char* {
        char* p = ws + off;
        off += (bytes + 255) & ~(size_t)255;
        return p;
    };
    float*     xb   = (float*)    take((size_t)BT * CNUM * 4);
    _Float16*  h16  = (_Float16*) take((size_t)BT * CNUM * 2);
    _Float16*  q16  = (_Float16*) take((size_t)BT * CNUM * 2);
    _Float16*  k16  = (_Float16*) take((size_t)BT * CNUM * 2);
    _Float16*  v16  = (_Float16*) take((size_t)BT * CNUM * 2);
    _Float16*  y16  = (_Float16*) take((size_t)BT * CNUM * 2);
    _Float16*  m16  = (_Float16*) take((size_t)BT * 4 * CNUM * 2);
    _Float16*  wq16 = (_Float16*) take((size_t)LNUM * CNUM * CNUM * 2);
    _Float16*  wk16 = (_Float16*) take((size_t)LNUM * CNUM * CNUM * 2);
    _Float16*  wv16 = (_Float16*) take((size_t)LNUM * CNUM * CNUM * 2);
    _Float16*  wo16 = (_Float16*) take((size_t)LNUM * CNUM * CNUM * 2);
    _Float16*  w116 = (_Float16*) take((size_t)LNUM * CNUM * 4 * CNUM * 2);
    _Float16*  w216 = (_Float16*) take((size_t)LNUM * 4 * CNUM * CNUM * 2);

    const int nproj = LNUM * CNUM * CNUM;        // 2M
    const int nmlp  = LNUM * CNUM * 4 * CNUM;    // 8M
    k_cvt_f32_f16<<<2048, 256, 0, stream>>>(wq, wq16, nproj);
    k_cvt_f32_f16<<<2048, 256, 0, stream>>>(wk, wk16, nproj);
    k_cvt_f32_f16<<<2048, 256, 0, stream>>>(wv, wv16, nproj);
    k_cvt_f32_f16<<<2048, 256, 0, stream>>>(wo, wo16, nproj);
    k_cvt_f32_f16<<<4096, 256, 0, stream>>>(w1, w116, nmlp);
    k_cvt_f32_f16<<<4096, 256, 0, stream>>>(w2, w216, nmlp);

    hipMemcpyAsync(xb, x, (size_t)BT * CNUM * 4, hipMemcpyDeviceToDevice, stream);

    const dim3 blk(256);
    const dim3 gLN(BT);
    const dim3 gP(CNUM / 128, BT / 128);             // (8, 32)   N=C
    const dim3 gM1(4 * CNUM / 128, BT / 128);        // (32, 32)  N=4C
    const dim3 gAtt(BNUM * HNUM * (TNUM / 128));     // 512

    for (int i = 0; i < LNUM; ++i) {
        const size_t wof = (size_t)i * CNUM * CNUM;
        const size_t mof = (size_t)i * CNUM * 4 * CNUM;

        k_layernorm<<<gLN, blk, 0, stream>>>(xb, ln1_g + i * CNUM, ln1_b + i * CNUM, h16, nullptr);

        k_gemm<<<gP, blk, 0, stream>>>(h16, wq16 + wof, bq + i * CNUM, q16, nullptr, nullptr,
                                       BT, CNUM, CNUM, GEMM_MODE_F16);
        k_gemm<<<gP, blk, 0, stream>>>(h16, wk16 + wof, bk + i * CNUM, k16, nullptr, nullptr,
                                       BT, CNUM, CNUM, GEMM_MODE_F16);
        k_gemm<<<gP, blk, 0, stream>>>(h16, wv16 + wof, bv + i * CNUM, v16, nullptr, nullptr,
                                       BT, CNUM, CNUM, GEMM_MODE_F16);

        k_attention<<<gAtt, blk, 0, stream>>>(q16, k16, v16, y16);

        k_gemm<<<gP, blk, 0, stream>>>(y16, wo16 + wof, bo + i * CNUM, nullptr, xb, xb,
                                       BT, CNUM, CNUM, GEMM_MODE_RESID);

        k_layernorm<<<gLN, blk, 0, stream>>>(xb, ln2_g + i * CNUM, ln2_b + i * CNUM, h16, nullptr);

        k_gemm<<<gM1, blk, 0, stream>>>(h16, w116 + mof, b1 + i * 4 * CNUM, m16, nullptr, nullptr,
                                        BT, 4 * CNUM, CNUM, GEMM_MODE_GELU);
        k_gemm<<<gP, blk, 0, stream>>>(m16, w216 + mof, b2 + i * CNUM, nullptr, xb, xb,
                                       BT, CNUM, 4 * CNUM, GEMM_MODE_RESID);
    }

    k_layernorm<<<gLN, blk, 0, stream>>>(xb, lnf_g, lnf_b, nullptr, (float*)d_out);
}
